// WikiHMM_65996467471164
// MI455X (gfx1250) — matrix operand order repaired and checked by reading, not compiled, not run on hardware
//
#include <hip/hip_runtime.h>

#define H     512
#define T     50003
#define BATCH 32
#define SEQ   256
#define EPS   1e-10f

typedef __attribute__((ext_vector_type(16))) _Float16 v16h;
typedef __attribute__((ext_vector_type(8)))  float    v8f;

// ---- workspace layout (float offsets) ----
#define WS_LOGPI 0
#define WS_ENT   512
#define WS_LOGZ  1024
#define WS_SCAL  1536                 // [0]=entropy_loss, [1]=colsum_reg
#define WS_ALPHA 2048                 // 2 * 32 * 512 (double buffer)
#define WS_ASWZ  34816                // 512*512 halves = 131072 floats
#define WS_OBS   165888               // 256*32*512 floats

// ---- output layout (float offsets) ----
#define OUT_PI   0
#define OUT_A    512
#define OUT_B    (512 + 512*512)
#define OUT_LOSS (512 + 512*512 + 512*50003)

// ---------------- pi softmax ----------------
__global__ __launch_bounds__(512) void k_pi(const float* __restrict__ logits,
                                            float* __restrict__ out_pi,
                                            float* __restrict__ log_pi) {
  __shared__ float red[512];
  int tid = threadIdx.x;
  float x = logits[tid];
  red[tid] = x; __syncthreads();
  for (int off = 256; off >= 1; off >>= 1) {
    if (tid < off) red[tid] = fmaxf(red[tid], red[tid + off]);
    __syncthreads();
  }
  float M = red[0]; __syncthreads();
  float e = __expf(x - M);
  red[tid] = e; __syncthreads();
  for (int off = 256; off >= 1; off >>= 1) {
    if (tid < off) red[tid] += red[tid + off];
    __syncthreads();
  }
  float p = e / red[0];
  out_pi[tid] = p;
  log_pi[tid] = __logf(p + EPS);
}

// ---------------- transition softmax + entropy + f16 WMMA-swizzled A' ----------------
// B-operand layout used here (contiguous K, matching A in k_scan; 32x16 KxN): lane = ha*16 + (N%16),
// element e: K = kt*32 + ha*16 + e.  Stored as v16h frags: [(kt*32+nt)*32 + lane].
__global__ __launch_bounds__(256) void k_trans(const float* __restrict__ tl,
                                               float* __restrict__ out_A,
                                               float* __restrict__ ent,
                                               _Float16* __restrict__ aswz) {
  __shared__ float red[256];
  int i = blockIdx.x, tid = threadIdx.x;
  const float* row = tl + i * H;
  float x0 = row[tid], x1 = row[tid + 256];
  red[tid] = fmaxf(x0, x1); __syncthreads();
  for (int off = 128; off >= 1; off >>= 1) {
    if (tid < off) red[tid] = fmaxf(red[tid], red[tid + off]);
    __syncthreads();
  }
  float M = red[0]; __syncthreads();
  float e0 = __expf(x0 - M), e1 = __expf(x1 - M);
  red[tid] = e0 + e1; __syncthreads();
  for (int off = 128; off >= 1; off >>= 1) {
    if (tid < off) red[tid] += red[tid + off];
    __syncthreads();
  }
  float inv = 1.0f / red[0]; __syncthreads();
  float p0 = e0 * inv, p1 = e1 * inv;
  out_A[i * H + tid] = p0;
  out_A[i * H + tid + 256] = p1;
  // swizzled f16 writes: i is the K index, j the N index
  int kt = i >> 5, rr = i & 31, ha = rr >> 4, e = rr & 15;
  {
    int j = tid;
    aswz[((kt * 32 + (j >> 4)) * 32 + ha * 16 + (j & 15)) * 16 + e] = (_Float16)(p0 + EPS);
    j = tid + 256;
    aswz[((kt * 32 + (j >> 4)) * 32 + ha * 16 + (j & 15)) * 16 + e] = (_Float16)(p1 + EPS);
  }
  red[tid] = -(p0 * __logf(p0 + EPS) + p1 * __logf(p1 + EPS)); __syncthreads();
  for (int off = 128; off >= 1; off >>= 1) {
    if (tid < off) red[tid] += red[tid + off];
    __syncthreads();
  }
  if (tid == 0) ent[i] = red[0];
}

// ---------------- column-sum regularizer + entropy loss scalars ----------------
__global__ __launch_bounds__(512) void k_scalars(const float* __restrict__ out_A,
                                                 const float* __restrict__ ent,
                                                 const float* __restrict__ thr_p,
                                                 float* __restrict__ scal) {
  __shared__ float red[512];
  int tid = threadIdx.x;
  float s = 0.f;
  for (int i = 0; i < H; ++i) s += out_A[i * H + tid];
  float thr = thr_p[0];
  red[tid] = fmaxf(thr - s, 0.f); __syncthreads();
  for (int off = 256; off >= 1; off >>= 1) {
    if (tid < off) red[tid] += red[tid + off];
    __syncthreads();
  }
  float colreg = red[0]; __syncthreads();
  red[tid] = ent[tid]; __syncthreads();
  for (int off = 256; off >= 1; off >>= 1) {
    if (tid < off) red[tid] += red[tid + off];
    __syncthreads();
  }
  if (tid == 0) { scal[0] = (red[0] / (float)H) * 0.1f; scal[1] = colreg; }
}

// ---------------- emission softmax (memory-dominant) + per-row logZ ----------------
__global__ __launch_bounds__(256) void k_emis(const float* __restrict__ el,
                                              float* __restrict__ out_B,
                                              float* __restrict__ logZ) {
  __shared__ float red[256];
  int h = blockIdx.x, tid = threadIdx.x;
  const float* row = el + h * T;
  float lm = -3.402823466e38f;
  for (int t = tid; t < T; t += 256) lm = fmaxf(lm, row[t]);
  red[tid] = lm; __syncthreads();
  for (int off = 128; off >= 1; off >>= 1) {
    if (tid < off) red[tid] = fmaxf(red[tid], red[tid + off]);
    __syncthreads();
  }
  float M = red[0]; __syncthreads();
  float s = 0.f;
  for (int t = tid; t < T; t += 256) s += __expf(row[t] - M);
  red[tid] = s; __syncthreads();
  for (int off = 128; off >= 1; off >>= 1) {
    if (tid < off) red[tid] += red[tid + off];
    __syncthreads();
  }
  float sum = red[0];
  float inv = 1.0f / sum;
  float* orow = out_B + h * T;
  for (int t = tid; t < T; t += 256) orow[t] = __expf(row[t] - M) * inv;
  if (tid == 0) logZ[h] = M + __logf(sum);
}

// ---------------- gather per-token emission log-probs: obs[s][b][h] ----------------
__global__ __launch_bounds__(512) void k_obs(const int* __restrict__ text,
                                             const float* __restrict__ el,
                                             const float* __restrict__ logZ,
                                             float* __restrict__ obs) {
  int bs = blockIdx.x;          // = b*SEQ + s
  int b = bs >> 8, s = bs & 255;
  int tok = text[bs];
  int h = threadIdx.x;
  float lp = el[h * T + tok] - logZ[h];
  obs[(s * BATCH + b) * H + h] = __logf(__expf(lp) + EPS);
}

// ---------------- alpha0 = log_pi + obs[0] ----------------
__global__ __launch_bounds__(512) void k_init(const float* __restrict__ log_pi,
                                              const float* __restrict__ obs,
                                              float* __restrict__ alpha) {
  int b = blockIdx.x, j = threadIdx.x;
  alpha[b * H + j] = log_pi[j] + obs[b * H + j];
}

// ---------------- persistent 255-step forward scan, WMMA f16 ----------------
// Block mt owns batch rows [mt*16, mt*16+16): per-row recursion is independent,
// so only __syncthreads() between steps (no grid sync, no relaunch).
// kt loop kept rolled (#pragma unroll 2) so B-fragment loads cannot be hoisted
// out of the t loop (which previously caused 256-VGPR spill-to-scratch).
__global__ __launch_bounds__(512) void k_scan(const _Float16* __restrict__ aswz,
                                              const float* __restrict__ obs,
                                              float* __restrict__ alpha) {
  __shared__ alignas(32) _Float16 P[16 * 32 * 16];  // A-operand frags: [kt][lane][e]
  __shared__ float red[512];
  __shared__ float mx[16];
  const v16h* bsw = (const v16h*)aswz;
  const v16h* pv  = (const v16h*)P;
  int tid  = threadIdx.x;
  int lane = tid & 31;
  int w    = tid >> 5;            // wave 0..15 -> N-tiles 2w, 2w+1
  int mt   = blockIdx.x;          // 0..1
  int nt0  = w * 2, nt1 = w * 2 + 1;

  for (int t = 1; t < SEQ; ++t) {
    const float* aIn = alpha + ((t - 1) & 1) * BATCH * H;
    float*      aOut = alpha + (t & 1) * BATCH * H;

    // row maxes m_b over 512 for this block's 16 rows
    {
      int r = tid & 15, seg = tid >> 4;
      const float* rp = aIn + (mt * 16 + r) * H + seg * 16;
      float lm = rp[0];
      #pragma unroll
      for (int q = 1; q < 16; ++q) lm = fmaxf(lm, rp[q]);
      red[tid] = lm; __syncthreads();
      for (int off = 256; off >= 16; off >>= 1) {
        if (tid < off) red[tid] = fmaxf(red[tid], red[tid + off]);
        __syncthreads();
      }
      if (tid < 16) mx[tid] = red[tid];
      __syncthreads();
    }

    // build P = exp(alpha - m) in f16 A-operand layout:
    //   lane ha=ln>>4, m=ln&15; element e -> K = kt*32 + ha*16 + e
    for (int idx = tid; idx < 8192; idx += 512) {
      int kt = idx >> 9, q = idx & 511, ln = q >> 4, e = q & 15;
      int ha = ln >> 4, m = ln & 15;
      int k = kt * 32 + ha * 16 + e;
      P[idx] = (_Float16)__expf(aIn[(mt * 16 + m) * H + k] - mx[m]);
    }
    __syncthreads();

    // one kt loop, shared A-fragment, two independent WMMA accumulator chains
    v8f acc0 = {0.f, 0.f, 0.f, 0.f, 0.f, 0.f, 0.f, 0.f};
    v8f acc1 = {0.f, 0.f, 0.f, 0.f, 0.f, 0.f, 0.f, 0.f};
    #pragma unroll 2
    for (int kt = 0; kt < 16; ++kt) {
      v16h a  = pv[kt * 32 + lane];
      v16h b0 = bsw[(kt * 32 + nt0) * 32 + lane];
      v16h b1 = bsw[(kt * 32 + nt1) * 32 + lane];
      acc0 = __builtin_amdgcn_wmma_f32_16x16x32_f16(false, a, false, b0,
                                                    (short)0, acc0, false, false);
      acc1 = __builtin_amdgcn_wmma_f32_16x16x32_f16(false, a, false, b1,
                                                    (short)0, acc1, false, false);
    }

    // C/D layout: VGPR v -> row M = (lane/16)*8 + v, col N = lane%16
    const float* obst = obs + t * BATCH * H;
    int n0 = nt0 * 16 + (lane & 15);
    int n1 = nt1 * 16 + (lane & 15);
    int mb = (lane >> 4) * 8;
    #pragma unroll
    for (int v = 0; v < 8; ++v) {
      int b = mt * 16 + mb + v;
      float m = mx[mb + v];
      aOut[b * H + n0] = m + __logf(acc0[v]) + obst[b * H + n0];
      aOut[b * H + n1] = m + __logf(acc1[v]) + obst[b * H + n1];
    }
    __syncthreads();
  }
}

// ---------------- final logsumexp over states, mean, combine losses ----------------
__global__ __launch_bounds__(512) void k_final(const float* __restrict__ alpha,
                                               const float* __restrict__ scal,
                                               float* __restrict__ out_loss) {
  __shared__ float red[512];
  __shared__ float lls[BATCH];
  int tid = threadIdx.x;
  int b = tid >> 4, q = tid & 15;
  const float* af = alpha + ((SEQ - 1) & 1) * BATCH * H;
  const float* rp = af + b * H + q * 32;
  float lm = rp[0];
  #pragma unroll
  for (int jj = 1; jj < 32; ++jj) lm = fmaxf(lm, rp[jj]);
  red[tid] = lm; __syncthreads();
  for (int off = 8; off >= 1; off >>= 1) {
    if (q < off) red[tid] = fmaxf(red[tid], red[tid + off]);
    __syncthreads();
  }
  float M = red[b << 4]; __syncthreads();
  float s = 0.f;
  #pragma unroll
  for (int jj = 0; jj < 32; ++jj) s += __expf(rp[jj] - M);
  red[tid] = s; __syncthreads();
  for (int off = 8; off >= 1; off >>= 1) {
    if (q < off) red[tid] += red[tid + off];
    __syncthreads();
  }
  if (q == 0) lls[b] = M + __logf(red[tid]);
  __syncthreads();
  if (tid == 0) {
    float tot = 0.f;
    for (int bb = 0; bb < BATCH; ++bb) tot += lls[bb];
    out_loss[0] = -(tot / (float)BATCH) + scal[0] + scal[1];
  }
}

extern "C" void kernel_launch(void* const* d_in, const int* in_sizes, int n_in,
                              void* d_out, int out_size, void* d_ws, size_t ws_size,
                              hipStream_t stream) {
  (void)in_sizes; (void)n_in; (void)out_size; (void)ws_size;
  const int*   text = (const int*)d_in[0];
  const float* il   = (const float*)d_in[1];
  const float* tl   = (const float*)d_in[2];
  const float* el   = (const float*)d_in[3];
  const float* thr  = (const float*)d_in[4];
  float* out = (float*)d_out;
  float* W   = (float*)d_ws;

  float*    log_pi = W + WS_LOGPI;
  float*    ent    = W + WS_ENT;
  float*    logZ   = W + WS_LOGZ;
  float*    scal   = W + WS_SCAL;
  float*    alpha  = W + WS_ALPHA;
  _Float16* aswz   = (_Float16*)(W + WS_ASWZ);
  float*    obs    = W + WS_OBS;

  k_pi     <<<1,          512, 0, stream>>>(il, out + OUT_PI, log_pi);
  k_trans  <<<H,          256, 0, stream>>>(tl, out + OUT_A, ent, aswz);
  k_scalars<<<1,          512, 0, stream>>>(out + OUT_A, ent, thr, scal);
  k_emis   <<<H,          256, 0, stream>>>(el, out + OUT_B, logZ);
  k_obs    <<<BATCH*SEQ,  512, 0, stream>>>(text, el, logZ, obs);
  k_init   <<<BATCH,      512, 0, stream>>>(log_pi, obs, alpha);
  k_scan   <<<2,          512, 0, stream>>>(aswz, obs, alpha);
  k_final  <<<1,          512, 0, stream>>>(alpha, scal, out + OUT_LOSS);
}
